// MambaVision3DMixer_83854941487463
// MI455X (gfx1250) — matrix operand order repaired and checked by reading, not compiled, not run on hardware
//
#include <hip/hip_runtime.h>
#include <hip/hip_bf16.h>

// ---------------------------------------------------------------------------
// MambaVision 3D mixer, fused for MI455X (gfx1250, wave32, WMMA).
// x: (1, 64, 96, 96, 96) fp32.
//
// Pipeline:
//   k_init       : zero the 128-float stats accumulator
//   k_stats      : per-channel sum / sumsq (f32 atomics)
//   k_finalize   : fold mean/var/gamma/beta -> per-channel scale/shift
//   k_inproj     : xn = x*scale+shift (f16), xp = Win*xn + b  [WMMA f16->f32]
//                  (transposed padded LDS tile -> B frags are 2x ds_load_b128)
//   k_dw  (x3)   : depthwise 3x3x3 (+SiLU) on f16 tensors, f32 accumulate
//   k_outproj    : y = Wout*[x1;x2] + b + residual            [WMMA f16->f32]
//                  (B tile staged with GLOBAL_LOAD_ASYNC_TO_LDS_B128)
// ---------------------------------------------------------------------------

#define NSP   884736          // 96*96*96
#define DIMS  96
#define CCH   64
#define HALF  32
#define BPAD  72              // padded row stride (halves): 144 B = 9*16 B

typedef __attribute__((ext_vector_type(16))) _Float16 v16h;
typedef __attribute__((ext_vector_type(8)))  float    v8f;
typedef __attribute__((ext_vector_type(4)))  int      v4i;

typedef __attribute__((address_space(1))) v4i* gv4i_p;   // global int4*
typedef __attribute__((address_space(3))) v4i* lv4i_p;   // LDS int4*

#if __has_builtin(__builtin_amdgcn_global_load_async_to_lds_b128) && \
    __has_builtin(__builtin_amdgcn_s_wait_asynccnt)
#define HAVE_ASYNC_LDS 1
#else
#define HAVE_ASYNC_LDS 0
#endif

// ---------------------------------------------------------------- stats ----
__global__ void k_init(float* stats) {
  if (threadIdx.x < 128) stats[threadIdx.x] = 0.f;
}

__global__ void k_stats(const float* __restrict__ x, float* __restrict__ stats) {
  const int c = blockIdx.y;
  const size_t base = (size_t)c * NSP + (size_t)blockIdx.x * (256 * 64);
  const float* p = x + base + threadIdx.x;
  float s = 0.f, s2 = 0.f;
  #pragma unroll 8
  for (int i = 0; i < 64; ++i) { float v = p[(size_t)i * 256]; s += v; s2 += v * v; }
  for (int off = 16; off > 0; off >>= 1) {
    s  += __shfl_down(s,  off);
    s2 += __shfl_down(s2, off);
  }
  __shared__ float red[16];
  const int lane = threadIdx.x & 31, wv = threadIdx.x >> 5;
  if (lane == 0) { red[wv] = s; red[8 + wv] = s2; }
  __syncthreads();
  if (threadIdx.x == 0) {
    float a = 0.f, b = 0.f;
    #pragma unroll
    for (int i = 0; i < 8; ++i) { a += red[i]; b += red[8 + i]; }
    atomicAdd(&stats[c], a);
    atomicAdd(&stats[64 + c], b);
  }
}

__global__ void k_finalize(const float* __restrict__ stats,
                           const float* __restrict__ nw,
                           const float* __restrict__ nb,
                           float* __restrict__ ss) {
  const int c = threadIdx.x;            // 64 threads
  const float inv_n = 1.f / (float)NSP;
  const float mean = stats[c] * inv_n;
  const float var  = stats[64 + c] * inv_n - mean * mean;
  const float r    = rsqrtf(var + 1e-5f);
  const float sc   = nw[c] * r;
  ss[c]      = sc;                      // scale
  ss[64 + c] = nb[c] - mean * sc;       // shift
}

// ----------------------------------------------------- 1x1 conv via WMMA ---
// 256 threads = 8 waves, 128 contiguous voxels per block.
// Activation tile staged TRANSPOSED: lBt[spatial 0..127][channel 0..63 +pad],
// row stride 72 halves (144 B, always 16-B aligned) -> each B fragment is two
// aligned ds_load_b128 per lane.
__global__ void k_inproj(const float* __restrict__ x,
                         const float* __restrict__ w,      // [o][c] row-major
                         const float* __restrict__ bias,
                         const float* __restrict__ ss,     // scale[64], shift[64]
                         _Float16* __restrict__ xp) {
  __shared__ __align__(16) _Float16 lA[CCH * CCH];     // 8 KB  weights f16
  __shared__ __align__(16) _Float16 lBt[128 * BPAD];   // 18 KB transposed acts
  const int t  = threadIdx.x;
  const int p0 = blockIdx.x * 128;

  // stage A (weights) -> f16
  for (int i = t; i < CCH * CCH; i += 256) lA[i] = (_Float16)w[i];

  // stage B transposed: thread owns spatial p = t&127, channels chalf*32..+31.
  // Global reads: per channel, 32 lanes x 4 B contiguous (coalesced).
  // LDS writes: 4x ds_store_b128, bank mapping 36*p mod 64 (bijective / 16 lanes).
  {
    const int p   = t & 127;
    const int ch0 = (t >> 7) * 32;          // wave-uniform
    const float* src = x + (size_t)ch0 * NSP + p0 + p;
    union { _Float16 h[32]; uint4 q[4]; } v;
    #pragma unroll
    for (int j = 0; j < 32; ++j) {
      const float f = src[(size_t)j * NSP];
      v.h[j] = (_Float16)(f * ss[ch0 + j] + ss[64 + ch0 + j]);
    }
    _Float16* dst = &lBt[p * BPAD + ch0];
    #pragma unroll
    for (int i = 0; i < 4; ++i) *(uint4*)(dst + i * 8) = v.q[i];
  }
  __syncthreads();

  const int wave = t >> 5;
  const int lane = t & 31;
  const int col  = lane & 15;        // N index / A row-in-tile
  const int hi   = lane >> 4;        // lane-half selects K sub-range
  const int pw   = wave * 16;
  const int kb   = hi * 16;

  // B fragments: elems 0..7 -> K = 8*hi + i, elems 8..15 -> K = 16 + 8*hi + i -> 2x b128 each
  union { v16h v; uint4 q[2]; } ub0, ub1;
  {
    const _Float16* bp = &lBt[(pw + col) * BPAD];
    ub0.q[0] = *(const uint4*)(bp + hi * 8);
    ub0.q[1] = *(const uint4*)(bp + 16 + hi * 8);
    ub1.q[0] = *(const uint4*)(bp + 32 + hi * 8);
    ub1.q[1] = *(const uint4*)(bp + 48 + hi * 8);
  }

  const size_t pout = (size_t)p0 + pw + col;
  #pragma unroll
  for (int m = 0; m < 4; ++m) {
    const _Float16* arow = &lA[(m * 16 + col) * CCH];
    // A fragments: elems 0..7 -> K = 8*hi + i, elems 8..15 -> K = 16 + 8*hi + i
    v16h a0, a1;
    #pragma unroll
    for (int i = 0; i < 8; ++i) {
      a0[i]     = arow[hi * 8 + i];
      a0[8 + i] = arow[16 + hi * 8 + i];
      a1[i]     = arow[32 + hi * 8 + i];
      a1[8 + i] = arow[48 + hi * 8 + i];
    }
    v8f acc;
    #pragma unroll
    for (int r = 0; r < 8; ++r) acc[r] = bias[m * 16 + hi * 8 + r];

    acc = __builtin_amdgcn_wmma_f32_16x16x32_f16(false, a0, false, ub0.v,
                                                 (short)0, acc, false, false);
    acc = __builtin_amdgcn_wmma_f32_16x16x32_f16(false, a1, false, ub1.v,
                                                 (short)0, acc, false, false);
    #pragma unroll
    for (int r = 0; r < 8; ++r)
      xp[(size_t)(m * 16 + hi * 8 + r) * NSP + pout] = (_Float16)acc[r];
  }
}

// ------------------------------------------------- depthwise 3x3x3 (+SiLU) -
#define TD 2
#define TH 4
#define TW 32

__global__ void k_dw(const _Float16* __restrict__ in,   // [nch][NSP]
                     const float* __restrict__ wconv,   // [nch][27]
                     const float* __restrict__ bconv,   // [nch]
                     _Float16* __restrict__ out,
                     int applySilu) {
  const int c = blockIdx.y;
  int bx = blockIdx.x;
  const int wt = bx % 3;  bx /= 3;
  const int ht = bx % 24; bx /= 24;
  const int dt = bx;                       // 0..47
  const int w0 = wt * TW, h0 = ht * TH, d0 = dt * TD;

  __shared__ float tile[(TD + 2) * (TH + 2) * (TW + 2)];   // 816 f32
  __shared__ float wk[27];

  const _Float16* src = in + (size_t)c * NSP;
  for (int i = threadIdx.x; i < (TD + 2) * (TH + 2) * (TW + 2); i += 256) {
    const int lw = i % (TW + 2);
    int r = i / (TW + 2);
    const int lh = r % (TH + 2);
    const int ld = r / (TH + 2);
    const int gw = w0 + lw - 1, gh = h0 + lh - 1, gd = d0 + ld - 1;
    float v = 0.f;
    if ((unsigned)gw < (unsigned)DIMS && (unsigned)gh < (unsigned)DIMS &&
        (unsigned)gd < (unsigned)DIMS)
      v = (float)src[((size_t)gd * DIMS + gh) * DIMS + gw];
    tile[i] = v;
  }
  if (threadIdx.x < 27) wk[threadIdx.x] = wconv[c * 27 + threadIdx.x];
  __syncthreads();

  const int lw = threadIdx.x % TW;
  const int lh = (threadIdx.x / TW) % TH;
  const int ld = threadIdx.x / (TW * TH);

  float acc = bconv[c];
  #pragma unroll
  for (int kd = 0; kd < 3; ++kd)
    #pragma unroll
    for (int kh = 0; kh < 3; ++kh)
      #pragma unroll
      for (int kw = 0; kw < 3; ++kw)
        acc += wk[(kd * 3 + kh) * 3 + kw] *
               tile[((ld + kd) * (TH + 2) + (lh + kh)) * (TW + 2) + (lw + kw)];
  if (applySilu) acc = acc / (1.f + __expf(-acc));
  out[(size_t)c * NSP + ((size_t)(d0 + ld) * DIMS + (h0 + lh)) * DIMS + (w0 + lw)] =
      (_Float16)acc;
}

// ------------------------------------------------ out-proj WMMA + residual -
// B tile is a straight f16 copy from workspace -> use async global->LDS DMA
// (ASYNCcnt), one b128 (int4) per lane per call.
__global__ void k_outproj(const _Float16* __restrict__ x1b,  // [32][NSP]
                          const _Float16* __restrict__ t2,   // [32][NSP]
                          const float* __restrict__ w,       // [o][c]
                          const float* __restrict__ bias,
                          const float* __restrict__ xres,    // residual (f32)
                          float* __restrict__ y) {
  __shared__ __align__(16) _Float16 lA[CCH * CCH];
  __shared__ __align__(16) _Float16 lB[CCH * 128];
  const int t  = threadIdx.x;
  const int p0 = blockIdx.x * 128;

  for (int i = t; i < CCH * CCH; i += 256) lA[i] = (_Float16)w[i];

  {
    const int row = t >> 2;
    const int ch  = (t & 3) * 32;
    const _Float16* src = (row < HALF)
        ? x1b + (size_t)row * NSP + p0 + ch
        : t2 + (size_t)(row - HALF) * NSP + p0 + ch;
    _Float16* dst = &lB[row * 128 + ch];
#if HAVE_ASYNC_LDS
    #pragma unroll
    for (int i = 0; i < 4; ++i)
      __builtin_amdgcn_global_load_async_to_lds_b128(
          (gv4i_p)(src + i * 8),
          (lv4i_p)(dst + i * 8),
          0, 0);
    __builtin_amdgcn_s_wait_asynccnt(0);
#else
    #pragma unroll
    for (int i = 0; i < 4; ++i)
      *(uint4*)(dst + i * 8) = *(const uint4*)(src + i * 8);
#endif
  }
  __syncthreads();

  const int wave = t >> 5;
  const int lane = t & 31;
  const int col  = lane & 15;
  const int hi   = lane >> 4;
  const int pw   = wave * 16;
  const int kb   = hi * 16;

  // conflict-free u16 gather (16 consecutive halves / 16 lanes, 2-lane bcast)
  v16h b0, b1;
  #pragma unroll
  for (int i = 0; i < 16; ++i) {
    b0[i] = lB[(kb + i) * 128 + pw + col];
    b1[i] = lB[(32 + kb + i) * 128 + pw + col];
  }

  const size_t pout = (size_t)p0 + pw + col;
  #pragma unroll
  for (int m = 0; m < 4; ++m) {
    const _Float16* arow = &lA[(m * 16 + col) * CCH];
    v16h a0, a1;
    #pragma unroll
    for (int i = 0; i < 8; ++i) {
      a0[i]     = arow[kb + i];
      a0[8 + i] = arow[kb + 8 + i];
      a1[i]     = arow[32 + kb + i];
      a1[8 + i] = arow[32 + kb + 8 + i];
    }
    v8f acc;
    #pragma unroll
    for (int r = 0; r < 8; ++r) acc[r] = bias[m * 16 + hi * 8 + r];

    acc = __builtin_amdgcn_wmma_f32_16x16x32_f16(false, a0, false, b0,
                                                 (short)0, acc, false, false);
    acc = __builtin_amdgcn_wmma_f32_16x16x32_f16(false, a1, false, b1,
                                                 (short)0, acc, false, false);
    #pragma unroll
    for (int r = 0; r < 8; ++r) {
      const size_t idx = (size_t)(m * 16 + hi * 8 + r) * NSP + pout;
      y[idx] = acc[r] + xres[idx];
    }
  }
}

// ---------------------------------------------------------------- launch ---
extern "C" void kernel_launch(void* const* d_in, const int* in_sizes, int n_in,
                              void* d_out, int out_size, void* d_ws, size_t ws_size,
                              hipStream_t stream) {
  const float* x          = (const float*)d_in[0];
  const float* norm_w     = (const float*)d_in[1];
  const float* norm_b     = (const float*)d_in[2];
  const float* in_proj_w  = (const float*)d_in[3];
  const float* in_proj_b  = (const float*)d_in[4];
  const float* conv_ssm_w = (const float*)d_in[5];
  const float* conv_ssm_b = (const float*)d_in[6];
  const float* mamba_w    = (const float*)d_in[7];
  const float* mamba_b    = (const float*)d_in[8];
  const float* conv_sym_w = (const float*)d_in[9];
  const float* conv_sym_b = (const float*)d_in[10];
  const float* out_proj_w = (const float*)d_in[11];
  const float* out_proj_b = (const float*)d_in[12];
  float* y = (float*)d_out;

  // workspace layout
  float* stats = (float*)d_ws;                              // 128 f32
  float* ss    = stats + 128;                               // scale/shift 128 f32
  _Float16* xp  = (_Float16*)((char*)d_ws + 1024);          // 64 x NSP  f16
  _Float16* t1  = xp + (size_t)CCH * NSP;                   // 32 x NSP  f16
  _Float16* t2  = t1 + (size_t)HALF * NSP;                  // 32 x NSP  f16
  _Float16* x1b = xp;  // xp is dead after both first-stage depthwise convs

  // 1) instance-norm statistics
  k_init<<<1, 128, 0, stream>>>(stats);
  k_stats<<<dim3(54, CCH), 256, 0, stream>>>(x, stats);
  k_finalize<<<1, 64, 0, stream>>>(stats, norm_w, norm_b, ss);

  // 2) norm + in_proj GEMM (WMMA)
  k_inproj<<<NSP / 128, 256, 0, stream>>>(x, in_proj_w, in_proj_b, ss, xp);

  // 3) depthwise branches
  const dim3 dwg(3 * 24 * 48, HALF);
  k_dw<<<dwg, 256, 0, stream>>>(xp, conv_ssm_w, conv_ssm_b, t1, 1);
  k_dw<<<dwg, 256, 0, stream>>>(xp + (size_t)HALF * NSP, conv_sym_w, conv_sym_b,
                                t2, 1);
  k_dw<<<dwg, 256, 0, stream>>>(t1, mamba_w, mamba_b, x1b, 0);

  // 4) out_proj GEMM (WMMA) + bias + residual
  k_outproj<<<NSP / 128, 256, 0, stream>>>(x1b, t2, out_proj_w, out_proj_b, x, y);
}
